// DualTowerConditionalBridge_77799037600232
// MI455X (gfx1250) — matrix operand order repaired and checked by reading, not compiled, not run on hardware
//
#include <hip/hip_runtime.h>
#include <hip/hip_bf16.h>
#include <math.h>

// ---------------------------------------------------------------------------
// DualTowerConditionalBridge on MI455X (gfx1250, wave32, WMMA + TDM)
//   B=2 LX=1024 DIM=2048 KV=1024 H=16 GT=8 LY=4096 HD=128 KHD=64
// Heavy GEMMs: TDM-staged LDS tiles + 2x2 register-blocked WMMA.
// Attention: flash-style WMMA with online softmax.
// ---------------------------------------------------------------------------

#define Bc   2
#define LXc  1024
#define DIMc 2048
#define KVc  1024
#define Hc   16
#define GTc  8
#define LYc  4096
#define HDc  128
#define KHDc 64
#define BTc  16   // B*GT
#define Sc   512  // GH*GW

typedef __attribute__((ext_vector_type(16))) _Float16 v16h;
typedef __attribute__((ext_vector_type(8)))  float    v8f;
typedef __attribute__((ext_vector_type(4)))  unsigned int v4u;
typedef __attribute__((ext_vector_type(8)))  int v8i_t;
typedef __attribute__((ext_vector_type(4)))  int v4i_t;

union AFrag { v16h v; _Float16 e[16]; };
union CFrag { v8f  v; float    e[8];  };

// A-matrix (16x32 f16) k index for element i, half-wave g  (ISA 7.12.2)
__device__ __forceinline__ int ka_off(int i, int g) {
  return (i < 8) ? (i + 8 * g) : (i + 8 + 8 * g);
}
// B-matrix (32x16 f16) k index: same k order as A (ka_off)
__device__ __forceinline__ int kb_off(int i, int g) { return ka_off(i, g); }

// ---------------------------------------------------------------------------
// Tensor Data Mover: 2D tile load global->LDS (ISA ch.8 D# descriptor).
// data_size=4B. Optional LDS row padding (pad_interval/pad_amount codes).
// Issue from ONE wave; fence with s_wait_tensorcnt; others sync via barrier.
// This toolchain's builtin takes 6 args (g0,g1,g2,g3,g2b,cpol).
// ---------------------------------------------------------------------------
__device__ __forceinline__ void
tdm_load_2d(unsigned lds_addr, const float* gsrc,
            unsigned tensor_d0, unsigned tensor_d1,
            unsigned tile_d0, unsigned tile_d1,
            unsigned long long stride_elems,
            int pad_enable, unsigned pad_interval_code, unsigned pad_amount_code)
{
  unsigned long long ga = (unsigned long long)(size_t)gsrc;
  // --- D# group 0 (128b): count=1 | lds_addr | global_addr[120:64] | type=2
  v4u g0;
  g0[0] = 1u;                                    // count=1, is_restore=0
  g0[1] = lds_addr;                              // LDS byte address
  g0[2] = (unsigned)(ga & 0xFFFFFFFFu);          // global_addr[95:64]
  g0[3] = (unsigned)((ga >> 32) & 0x01FFFFFFu)   // global_addr[120:96]
        | (2u << 30);                            // type=2 ("image")
  // --- D# group 1 (256b)
  unsigned w0 = (2u << 16);                      // data_size: 2 -> 4 bytes
  if (pad_enable)
    w0 |= (1u << 20) | (pad_interval_code << 22) | (pad_amount_code << 25);
  v8i_t g1;
  g1[0] = (int)w0;
  g1[1] = (int)((tensor_d0 & 0xFFFFu) << 16);                    // dim0 lo16
  g1[2] = (int)(((tensor_d0 >> 16) & 0xFFFFu) |
                ((tensor_d1 & 0xFFFFu) << 16));                  // dim0 hi16 | dim1 lo16
  g1[3] = (int)(((tensor_d1 >> 16) & 0xFFFFu) |
                ((tile_d0 & 0xFFFFu) << 16));                    // dim1 hi16 | tile_dim0
  g1[4] = (int)(tile_d1 & 0xFFFFu);                              // tile_dim1; tile_dim2=0
  g1[5] = (int)(unsigned)(stride_elems & 0xFFFFFFFFull);         // dim0_stride[31:0]
  g1[6] = (int)(unsigned)((stride_elems >> 32) & 0xFFFFull);     // dim0_stride[47:32]
  g1[7] = 0;
  v4i_t z4 = {0, 0, 0, 0};
  v8i_t z8 = {0, 0, 0, 0, 0, 0, 0, 0};
  __builtin_amdgcn_tensor_load_to_lds(g0, g1, z4, z4, z8, 0);
}

// ---------------------------------------------------------------------------
// TDM-staged, register-blocked WMMA GEMM:
//   C[m,n] = sum_k A[m,k] * W[n,k] + bias[n]
// Block tile 64x128 (8 waves as 2x4), wave tile 32x32 (2x2 WMMA frags).
// A/B K-chunks of 32 staged to LDS by the Tensor Data Mover with a 1-DWORD
// pad every 32 DWORDs -> 33-float row stride (bank-conflict-free reads).
// Requires M%64==0, N%128==0, K%32==0.
// ---------------------------------------------------------------------------
__global__ void __launch_bounds__(256)
gemm_wmma_tdm(const float* __restrict__ A, const float* __restrict__ W,
              const float* __restrict__ bias, float* __restrict__ C,
              int M, int N, int K)
{
  __shared__ float sA[64 * 33];
  __shared__ float sB[128 * 33];
  int tid = threadIdx.x;
  int lane = tid & 31;
  int wv = tid >> 5;
  int g = lane >> 4, lr = lane & 15;
  int nblk = N >> 7;
  int tileM = blockIdx.x / nblk;
  int tileN = blockIdx.x % nblk;
  int waveM = wv >> 2, waveN = wv & 3;   // 2 x 4 wave grid

  CFrag acc[2][2];
#pragma unroll
  for (int mi = 0; mi < 2; ++mi)
#pragma unroll
    for (int ni = 0; ni < 2; ++ni)
#pragma unroll
      for (int r = 0; r < 8; ++r) acc[mi][ni].e[r] = 0.0f;

  unsigned ldsA = (unsigned)(unsigned long long)(size_t)&sA[0];
  unsigned ldsB = (unsigned)(unsigned long long)(size_t)&sB[0];
  const float* Abase = A + (size_t)tileM * 64 * K;
  const float* Wbase = W + (size_t)tileN * 128 * K;

  for (int kb = 0; kb < K; kb += 32) {
    __syncthreads();                       // previous chunk's LDS reads done
    if (tid < 32) {                        // wave 0 drives the TDM
      tdm_load_2d(ldsA, Abase + kb, (unsigned)K, 64u, 32u, 64u,
                  (unsigned long long)K, 1, 4u /*32 dwords*/, 0u /*1 dword*/);
      tdm_load_2d(ldsB, Wbase + kb, (unsigned)K, 128u, 32u, 128u,
                  (unsigned long long)K, 1, 4u, 0u);
      __builtin_amdgcn_s_wait_tensorcnt(0);
    }
    __syncthreads();                       // tiles visible to all waves

    AFrag af[2], bf[2];
#pragma unroll
    for (int mi = 0; mi < 2; ++mi) {
      const float* ar = &sA[(waveM * 32 + mi * 16 + lr) * 33];
#pragma unroll
      for (int i = 0; i < 16; ++i) af[mi].e[i] = (_Float16)ar[ka_off(i, g)];
    }
#pragma unroll
    for (int ni = 0; ni < 2; ++ni) {
      const float* br = &sB[(waveN * 32 + ni * 16 + lr) * 33];
#pragma unroll
      for (int i = 0; i < 16; ++i) bf[ni].e[i] = (_Float16)br[kb_off(i, g)];
    }
#pragma unroll
    for (int mi = 0; mi < 2; ++mi)
#pragma unroll
      for (int ni = 0; ni < 2; ++ni)
        acc[mi][ni].v = __builtin_amdgcn_wmma_f32_16x16x32_f16(
            false, af[mi].v, false, bf[ni].v, (short)0, acc[mi][ni].v,
            false, false);
  }

#pragma unroll
  for (int mi = 0; mi < 2; ++mi)
#pragma unroll
    for (int ni = 0; ni < 2; ++ni) {
      int n = tileN * 128 + waveN * 32 + ni * 16 + lr;
      float bn = bias ? bias[n] : 0.0f;
#pragma unroll
      for (int r = 0; r < 8; ++r) {
        int m = tileM * 64 + waveM * 32 + mi * 16 + r + 8 * g;
        C[(size_t)m * N + n] = acc[mi][ni].e[r] + bn;
      }
    }
}

// ---------------------------------------------------------------------------
// Fallback WMMA GEMM (one 16x16 tile per wave) for small M (pooled path).
// ---------------------------------------------------------------------------
__global__ void __launch_bounds__(256)
gemm_wmma_bias(const float* __restrict__ A, const float* __restrict__ W,
               const float* __restrict__ bias, float* __restrict__ C,
               int M, int N, int K)
{
  int wave = (int)((blockIdx.x * blockDim.x + threadIdx.x) >> 5);
  int lane = threadIdx.x & 31;
  int ntiles = N >> 4;
  int total  = (M >> 4) * ntiles;
  if (wave >= total) return;                    // wave-uniform
  int tm = wave / ntiles, tn = wave % ntiles;
  int g = lane >> 4, lr = lane & 15;

  const float* arow = A + (size_t)(tm * 16 + lr) * K;
  const float* wrow = W + (size_t)(tn * 16 + lr) * K;

  CFrag acc;
#pragma unroll
  for (int r = 0; r < 8; ++r) acc.e[r] = 0.0f;

  for (int kb = 0; kb < K; kb += 32) {
    __builtin_prefetch(arow + kb + 128, 0, 0);
    __builtin_prefetch(wrow + kb + 128, 0, 0);
    AFrag a, b;
    int a0 = kb + 8 * g, a1 = kb + 16 + 8 * g;
    int b0 = kb + 8 * g, b1 = kb + 16 + 8 * g;
#pragma unroll
    for (int i = 0; i < 8; ++i) {
      a.e[i]     = (_Float16)arow[a0 + i];
      a.e[i + 8] = (_Float16)arow[a1 + i];
    }
#pragma unroll
    for (int i = 0; i < 8; ++i) { b.e[i] = (_Float16)wrow[b0 + i]; b.e[i + 8] = (_Float16)wrow[b1 + i]; }
    acc.v = __builtin_amdgcn_wmma_f32_16x16x32_f16(false, a.v, false, b.v,
                                                   (short)0, acc.v, false, false);
  }

  float bn = bias ? bias[tn * 16 + lr] : 0.0f;
#pragma unroll
  for (int r = 0; r < 8; ++r) {
    int m = tm * 16 + r + 8 * g;
    C[(size_t)m * N + tn * 16 + lr] = acc.e[r] + bn;
  }
}

// ---------------------------------------------------------------------------
// Block reduction helper (blockDim.x == 256)
// ---------------------------------------------------------------------------
__device__ __forceinline__ float block_reduce_sum(float val, float* red) {
  int tid = threadIdx.x;
  red[tid] = val;
  __syncthreads();
  for (int s = 128; s > 0; s >>= 1) {
    if (tid < s) red[tid] += red[tid + s];
    __syncthreads();
  }
  float r = red[0];
  __syncthreads();
  return r;
}

__global__ void __launch_bounds__(256)
ln_kernel(const float* __restrict__ in, float* __restrict__ out,
          const float* __restrict__ w, const float* __restrict__ b,
          int width, float eps)
{
  __shared__ float red[256];
  int row = blockIdx.x, tid = threadIdx.x;
  const float* x = in + (size_t)row * width;
  float s = 0.0f;
  for (int c = tid; c < width; c += 256) s += x[c];
  float mean = block_reduce_sum(s, red) / (float)width;
  float v = 0.0f;
  for (int c = tid; c < width; c += 256) { float d = x[c] - mean; v += d * d; }
  float var = block_reduce_sum(v, red) / (float)width;
  float rstd = rsqrtf(var + eps);
  for (int c = tid; c < width; c += 256)
    out[(size_t)row * width + c] = (x[c] - mean) * rstd * w[c] + b[c];
}

__global__ void __launch_bounds__(256)
modx_kernel(const float* __restrict__ x, const float* __restrict__ temb,
            float* __restrict__ out)
{
  __shared__ float red[256];
  int row = blockIdx.x, tid = threadIdx.x;
  const float* xr = x + (size_t)row * DIMc;
  float s = 0.0f;
  for (int c = tid; c < DIMc; c += 256) s += xr[c];
  float mean = block_reduce_sum(s, red) * (1.0f / DIMc);
  float v = 0.0f;
  for (int c = tid; c < DIMc; c += 256) { float d = xr[c] - mean; v += d * d; }
  float var = block_reduce_sum(v, red) * (1.0f / DIMc);
  float rstd = rsqrtf(var + 1e-5f);
  const float* tr = temb + (size_t)row * (2 * DIMc);
  for (int c = tid; c < DIMc; c += 256) {
    float ln = (xr[c] - mean) * rstd;
    out[(size_t)row * DIMc + c] = ln * (1.0f + tr[c]) + tr[DIMc + c];
  }
}

__global__ void __launch_bounds__(256)
rmsrope_kernel(float* __restrict__ x, const float* __restrict__ w,
               const float* __restrict__ cosb, const float* __restrict__ sinb)
{
  __shared__ float red[256];
  int row = blockIdx.x, tid = threadIdx.x;
  float* xr = x + (size_t)row * DIMc;
  float s = 0.0f;
  for (int c = tid; c < DIMc; c += 256) { float u = xr[c]; s += u * u; }
  float ms = block_reduce_sum(s, red) * (1.0f / DIMc);
  float rs = rsqrtf(ms + 1e-6f);
  const float* cr = cosb + (size_t)row * HDc;
  const float* sr = sinb + (size_t)row * HDc;
  for (int p = tid; p < (Hc * 64); p += 256) {
    int h = p >> 6, d = p & 63;
    int i1 = h * HDc + d, i2 = i1 + 64;
    float u1 = xr[i1] * rs * w[i1];
    float u2 = xr[i2] * rs * w[i2];
    xr[i1] = u1 * cr[d]      - u2 * sr[d];
    xr[i2] = u2 * cr[d + 64] + u1 * sr[d + 64];
  }
}

__global__ void __launch_bounds__(256)
interp_silu_kernel(const float* __restrict__ pooled, float* __restrict__ out)
{
  int row = blockIdx.x;            // b*LX + l
  int b = row >> 10, l = row & (LXc - 1);
  float coord = (l + 0.5f) * ((float)GTc / (float)LXc) - 0.5f;
  coord = fminf(fmaxf(coord, 0.0f), (float)(GTc - 1));
  int lo = (int)floorf(coord);
  int hi = min(lo + 1, GTc - 1);
  float wt = coord - (float)lo;
  const float* plo = pooled + (size_t)(b * GTc + lo) * KVc;
  const float* phi = pooled + (size_t)(b * GTc + hi) * KVc;
  for (int c = threadIdx.x; c < KVc; c += 256) {
    float val = plo[c] * (1.0f - wt) + phi[c] * wt;
    out[(size_t)row * KVc + c] = val / (1.0f + __expf(-val));
  }
}

__global__ void __launch_bounds__(256)
qp_kernel(const float* __restrict__ probe, const float* __restrict__ mha_in_w,
          const float* __restrict__ mha_in_b, float* __restrict__ qp)
{
  int n = blockIdx.x * 256 + threadIdx.x;
  if (n >= KVc) return;
  const float* wr = mha_in_w + (size_t)n * KVc;
  float s = mha_in_b[n];
  for (int k = 0; k < KVc; ++k) s += probe[k] * wr[k];
  qp[n] = s;
}

__global__ void __launch_bounds__(256)
pool_attn_kernel(const float* __restrict__ qp, const float* __restrict__ kp,
                 const float* __restrict__ vp, float* __restrict__ ctx)
{
  __shared__ float p[Sc];
  __shared__ float stats[2];
  int bt = blockIdx.x >> 4, h = blockIdx.x & 15;
  int tid = threadIdx.x;
  const float* q = qp + h * KHDc;
  for (int s = tid; s < Sc; s += 256) {
    const float* kr = kp + ((size_t)(bt * Sc + s)) * KVc + h * KHDc;
    float acc = 0.0f;
    for (int d = 0; d < KHDc; ++d) acc += q[d] * kr[d];
    p[s] = acc * 0.125f;
  }
  __syncthreads();
  if (tid == 0) {
    float mx = -1e30f;
    for (int s = 0; s < Sc; ++s) mx = fmaxf(mx, p[s]);
    stats[0] = mx;
  }
  __syncthreads();
  float mx = stats[0];
  for (int s = tid; s < Sc; s += 256) p[s] = __expf(p[s] - mx);
  __syncthreads();
  if (tid == 0) {
    float sum = 0.0f;
    for (int s = 0; s < Sc; ++s) sum += p[s];
    stats[1] = sum;
  }
  __syncthreads();
  if (tid < KHDc) {
    float acc = 0.0f;
    for (int s = 0; s < Sc; ++s)
      acc += p[s] * vp[((size_t)(bt * Sc + s)) * KVc + h * KHDc + tid];
    ctx[(size_t)bt * KVc + h * KHDc + tid] = acc / stats[1];
  }
}

// ---------------------------------------------------------------------------
// Flash attention: one wave per (b, h, 16-row Q tile). KV chunks of 32.
// ---------------------------------------------------------------------------
__global__ void __launch_bounds__(128)
flash_attn_kernel(const float* __restrict__ Q, const float* __restrict__ Kb,
                  const float* __restrict__ Vb, float* __restrict__ O)
{
  __shared__ float sS[4][16][33];
  __shared__ float rowm[4][16], rowl[4][16], rcorr[4][16];

  int w = threadIdx.x >> 5, lane = threadIdx.x & 31;
  int gw = blockIdx.x * 4 + w;
  int qt = gw & 63;
  int h  = (gw >> 6) & 15;
  int b  = gw >> 10;
  int g = lane >> 4, lr = lane & 15;

  if (lane < 16) { rowm[w][lane] = -1e30f; rowl[w][lane] = 0.0f; }

  const float qscale = 0.08838834764831845f;   // 1/sqrt(128)
  int m = qt * 16 + lr;
  const float* qrow = Q + ((size_t)(b * LXc + m)) * DIMc + h * HDc;

  AFrag qf[4];
#pragma unroll
  for (int c = 0; c < 4; ++c)
#pragma unroll
    for (int i = 0; i < 16; ++i)
      qf[c].e[i] = (_Float16)(qrow[c * 32 + ka_off(i, g)] * qscale);

  CFrag of[8];
#pragma unroll
  for (int t = 0; t < 8; ++t)
#pragma unroll
    for (int r = 0; r < 8; ++r) of[t].e[r] = 0.0f;

  for (int kv = 0; kv < LYc; kv += 32) {
#pragma unroll
    for (int nt = 0; nt < 2; ++nt) {
      CFrag sacc;
#pragma unroll
      for (int r = 0; r < 8; ++r) sacc.e[r] = 0.0f;
      const float* krow =
          Kb + ((size_t)(b * LYc + kv + nt * 16 + lr)) * DIMc + h * HDc;
#pragma unroll
      for (int c = 0; c < 4; ++c) {
        AFrag kf;
#pragma unroll
        for (int i = 0; i < 16; ++i)
          kf.e[i] = (_Float16)krow[c * 32 + kb_off(i, g)];
        sacc.v = __builtin_amdgcn_wmma_f32_16x16x32_f16(
            false, qf[c].v, false, kf.v, (short)0, sacc.v, false, false);
      }
#pragma unroll
      for (int r = 0; r < 8; ++r) sS[w][r + 8 * g][nt * 16 + lr] = sacc.e[r];
    }

    if (lane < 16) {
      int row = lane;
      float mx = rowm[w][row];
      float cmax = -1e30f;
      for (int j = 0; j < 32; ++j) cmax = fmaxf(cmax, sS[w][row][j]);
      float nm = fmaxf(mx, cmax);
      float corr = __expf(mx - nm);
      float sum = 0.0f;
      for (int j = 0; j < 32; ++j) {
        float pv = __expf(sS[w][row][j] - nm);
        sS[w][row][j] = pv;
        sum += pv;
      }
      rowl[w][row] = rowl[w][row] * corr + sum;
      rowm[w][row] = nm;
      rcorr[w][row] = corr;
    }

#pragma unroll
    for (int r = 0; r < 8; ++r) {
      float f = rcorr[w][r + 8 * g];
#pragma unroll
      for (int t = 0; t < 8; ++t) of[t].e[r] *= f;
    }

    AFrag pf;
#pragma unroll
    for (int i = 0; i < 16; ++i)
      pf.e[i] = (_Float16)sS[w][lr][ka_off(i, g)];

#pragma unroll
    for (int t = 0; t < 8; ++t) {
      AFrag vf;
#pragma unroll
      for (int i = 0; i < 16; ++i)
        vf.e[i] = (_Float16)
            Vb[((size_t)(b * LYc + kv + kb_off(i, g))) * DIMc + h * HDc +
               t * 16 + lr];
      of[t].v = __builtin_amdgcn_wmma_f32_16x16x32_f16(
          false, pf.v, false, vf.v, (short)0, of[t].v, false, false);
    }
  }

#pragma unroll
  for (int r = 0; r < 8; ++r) {
    int mm = qt * 16 + r + 8 * g;
    float inv = 1.0f / rowl[w][r + 8 * g];
#pragma unroll
    for (int t = 0; t < 8; ++t)
      O[((size_t)(b * LXc + mm)) * DIMc + h * HDc + t * 16 + lr] =
          of[t].e[r] * inv;
  }
}

// ---------------------------------------------------------------------------
// Host-side orchestration
// ---------------------------------------------------------------------------
static void launch_gemm(const float* A, const float* W, const float* bias,
                        float* C, int M, int N, int K, hipStream_t s) {
  if ((M % 64 == 0) && (N % 128 == 0)) {
    int blocks = (M / 64) * (N / 128);
    gemm_wmma_tdm<<<blocks, 256, 0, s>>>(A, W, bias, C, M, N, K);
  } else {
    int tiles = (M / 16) * (N / 16);
    int blocks = (tiles + 7) / 8;
    gemm_wmma_bias<<<blocks, 256, 0, s>>>(A, W, bias, C, M, N, K);
  }
}

// Workspace layout (bytes); lifetime-aliased, peak ~244 MB.
#define OFF_KP    ((size_t)0)
#define OFF_VP    ((size_t)33554432)
#define OFF_SILU  ((size_t)67108864)
#define OFF_TEMB  ((size_t)75497472)
#define OFF_XMOD  ((size_t)109051904)
#define OFF_YLN   ((size_t)125829120)
#define OFF_Q     ((size_t)0)
#define OFF_K     ((size_t)16777216)
#define OFF_V     ((size_t)159383552)
#define OFF_ATT   ((size_t)226492416)
#define OFF_QP    ((size_t)243269632)
#define OFF_CTX   (OFF_QP + 65536)
#define OFF_POOL  (OFF_QP + 131072)
#define OFF_POOLN (OFF_QP + 196608)

extern "C" void kernel_launch(void* const* d_in, const int* in_sizes, int n_in,
                              void* d_out, int out_size, void* d_ws,
                              size_t ws_size, hipStream_t stream) {
  (void)in_sizes; (void)n_in; (void)out_size; (void)ws_size;
  const float* x        = (const float*)d_in[0];
  const float* y        = (const float*)d_in[1];
  const float* x_cos    = (const float*)d_in[2];
  const float* x_sin    = (const float*)d_in[3];
  const float* y_cos    = (const float*)d_in[4];
  const float* y_sin    = (const float*)d_in[5];
  const float* probe    = (const float*)d_in[6];
  const float* mha_in_w = (const float*)d_in[7];
  const float* mha_in_b = (const float*)d_in[8];
  const float* mha_out_w= (const float*)d_in[9];
  const float* mha_out_b= (const float*)d_in[10];
  const float* pool_ln_w= (const float*)d_in[11];
  const float* pool_ln_b= (const float*)d_in[12];
  const float* ada_w    = (const float*)d_in[13];
  const float* ada_b    = (const float*)d_in[14];
  const float* ynorm_w  = (const float*)d_in[15];
  const float* ynorm_b  = (const float*)d_in[16];
  const float* q_w      = (const float*)d_in[17];
  const float* q_b      = (const float*)d_in[18];
  const float* k_w      = (const float*)d_in[19];
  const float* k_b      = (const float*)d_in[20];
  const float* v_w      = (const float*)d_in[21];
  const float* v_b      = (const float*)d_in[22];
  const float* o_w      = (const float*)d_in[23];
  const float* o_b      = (const float*)d_in[24];
  const float* rms_q_w  = (const float*)d_in[25];
  const float* rms_k_w  = (const float*)d_in[26];

  char* ws = (char*)d_ws;
  float* kp     = (float*)(ws + OFF_KP);
  float* vp     = (float*)(ws + OFF_VP);
  float* silu   = (float*)(ws + OFF_SILU);
  float* temb   = (float*)(ws + OFF_TEMB);
  float* xmod   = (float*)(ws + OFF_XMOD);
  float* yln    = (float*)(ws + OFF_YLN);
  float* qbuf   = (float*)(ws + OFF_Q);
  float* kbuf   = (float*)(ws + OFF_K);
  float* vbuf   = (float*)(ws + OFF_V);
  float* attout = (float*)(ws + OFF_ATT);
  float* qp     = (float*)(ws + OFF_QP);
  float* ctx    = (float*)(ws + OFF_CTX);
  float* pooled = (float*)(ws + OFF_POOL);
  float* pooln  = (float*)(ws + OFF_POOLN);
  float* out    = (float*)d_out;

  // ---- Phase 1: attention pooling over y ----
  qp_kernel<<<(KVc + 255) / 256, 256, 0, stream>>>(probe, mha_in_w, mha_in_b, qp);
  launch_gemm(y, mha_in_w + (size_t)KVc * KVc,     mha_in_b + KVc,     kp,
              BTc * Sc, KVc, KVc, stream);
  launch_gemm(y, mha_in_w + (size_t)2 * KVc * KVc, mha_in_b + 2 * KVc, vp,
              BTc * Sc, KVc, KVc, stream);
  pool_attn_kernel<<<BTc * Hc, 256, 0, stream>>>(qp, kp, vp, ctx);
  launch_gemm(ctx, mha_out_w, mha_out_b, pooled, BTc, KVc, KVc, stream);
  ln_kernel<<<BTc, 256, 0, stream>>>(pooled, pooln, pool_ln_w, pool_ln_b,
                                     KVc, 1e-6f);

  // ---- Phase 2: temporal embedding -> scale/shift ----
  interp_silu_kernel<<<Bc * LXc, 256, 0, stream>>>(pooln, silu);
  launch_gemm(silu, ada_w, ada_b, temb, Bc * LXc, 2 * DIMc, KVc, stream);

  // ---- Phase 3: modulate x; layernorm y ----
  modx_kernel<<<Bc * LXc, 256, 0, stream>>>(x, temb, xmod);
  ln_kernel<<<Bc * LYc, 256, 0, stream>>>(y, yln, ynorm_w, ynorm_b,
                                          KVc, 1e-6f);

  // ---- Phase 4: q/k/v projections + RMSNorm + RoPE ----
  launch_gemm(xmod, q_w, q_b, qbuf, Bc * LXc, DIMc, DIMc, stream);
  rmsrope_kernel<<<Bc * LXc, 256, 0, stream>>>(qbuf, rms_q_w, x_cos, x_sin);
  launch_gemm(yln, k_w, k_b, kbuf, Bc * LYc, DIMc, KVc, stream);
  rmsrope_kernel<<<Bc * LYc, 256, 0, stream>>>(kbuf, rms_k_w, y_cos, y_sin);
  launch_gemm(yln, v_w, v_b, vbuf, Bc * LYc, DIMc, KVc, stream);

  // ---- Phase 5: flash attention (WMMA) ----
  {
    int tiles = Bc * Hc * (LXc / 16);
    flash_attn_kernel<<<tiles / 4, 128, 0, stream>>>(qbuf, kbuf, vbuf, attout);
  }

  // ---- Phase 6: output projection ----
  launch_gemm(attout, o_w, o_b, out, Bc * LXc, DIMc, DIMc, stream);
}
